// MotifBondDecoder_79413945303068
// MI455X (gfx1250) — matrix operand order repaired and checked by reading, not compiled, not run on hardware
//
#include <hip/hip_runtime.h>

typedef __attribute__((ext_vector_type(16))) _Float16 v16h;
typedef __attribute__((ext_vector_type(8)))  float    v8f;
typedef __attribute__((ext_vector_type(4)))  unsigned int v4u;

#define HDIM   256   // shape embedding dim (K of the big GEMM)
#define HID    240   // hidden dim (N of the big GEMM)
#define IDD    64
#define CHD    16
#define POSD   32
// table row counts
#define NIDS   100
#define NCHG   8
#define NPOS   16
#define NTAB   (NIDS + NCHG + NPOS)   // 124 rows per side

// ---------------------------------------------------------------------------
// small helpers
// ---------------------------------------------------------------------------
__device__ __forceinline__ int find_seg(const int* __restrict__ starts, int nseg, int x) {
  // starts has nseg+1 entries; return s with starts[s] <= x < starts[s+1]
  int lo = 0, hi = nseg;
  while (hi - lo > 1) {
    int mid = (lo + hi) >> 1;
    if (starts[mid] <= x) lo = mid; else hi = mid;
  }
  return lo;
}

// ---------------------------------------------------------------------------
// 1) f32 -> f16 convert (shape embeddings)
// ---------------------------------------------------------------------------
__global__ void f32_to_f16_kernel(const float* __restrict__ src, _Float16* __restrict__ dst, int n) {
  int i = blockIdx.x * blockDim.x + threadIdx.x;
  if (i < n) dst[i] = (_Float16)src[i];
}

// 2) build transposed f16 copy of W1 rows [224,480): BT[n][k] = W1[224+k][n]
__global__ void build_bt_kernel(const float* __restrict__ W1, _Float16* __restrict__ BT) {
  int t = blockIdx.x * blockDim.x + threadIdx.x;
  if (t >= HID * HDIM) return;
  int n = t >> 8;          // 0..239
  int k = t & 255;         // 0..255
  BT[(size_t)n * HDIM + k] = (_Float16)W1[(size_t)(224 + k) * HID + n];
}

// 3) build the six small tables: T_{id,ch,pos}_{r,c} = emb @ W1-slice
//    Tr layout: [0,100*HID) id ; [100*HID,108*HID) charge ; [108*HID,124*HID) pos
__global__ void build_tables_kernel(const float* __restrict__ id_emb,
                                    const float* __restrict__ ch_emb,
                                    const float* __restrict__ pos_emb,
                                    const float* __restrict__ W1,
                                    float* __restrict__ Tr, float* __restrict__ Tc) {
  const int PER = NTAB * HID;
  int t = blockIdx.x * blockDim.x + threadIdx.x;
  if (t >= 2 * PER) return;
  int side  = t / PER;            // 0 = row endpoint, 1 = col endpoint
  int rem   = t - side * PER;
  int entry = rem / HID;
  int n     = rem - entry * HID;
  float* T  = side ? Tc : Tr;
  const float* emb; int K; int w1base; float* out;
  if (entry < NIDS) {
    emb = id_emb + (size_t)entry * IDD;  K = IDD;  w1base = 0;
    out = T + (size_t)entry * HID;
  } else if (entry < NIDS + NCHG) {
    int e2 = entry - NIDS;
    emb = ch_emb + (size_t)e2 * CHD;     K = CHD;  w1base = IDD;
    out = T + (size_t)(NIDS + e2) * HID;
  } else {
    int e2 = entry - (NIDS + NCHG);
    emb = pos_emb + (size_t)e2 * POSD;   K = POSD; w1base = IDD + CHD;
    out = T + (size_t)(NIDS + NCHG + e2) * HID;
  }
  w1base += side * 112;   // col-endpoint slice starts at W1 row 112
  float sum = 0.f;
  for (int k = 0; k < K; ++k)
    sum += emb[k] * W1[(size_t)(w1base + k) * HID + n];
  out[n] = sum;
}

// 4) single-block exclusive scan; writes starts[0..n] (starts[n] = total)
__global__ void scan_exclusive_kernel(const int* __restrict__ counts, int n, int* __restrict__ starts) {
  __shared__ int buf[1024];
  __shared__ int carry;
  if (threadIdx.x == 0) carry = 0;
  __syncthreads();
  for (int base = 0; base < n; base += 1024) {
    int i = base + (int)threadIdx.x;
    int v = (i < n) ? counts[i] : 0;
    buf[threadIdx.x] = v;
    __syncthreads();
    for (int off = 1; off < 1024; off <<= 1) {
      int t = (threadIdx.x >= (unsigned)off) ? buf[threadIdx.x - off] : 0;
      __syncthreads();
      buf[threadIdx.x] += t;
      __syncthreads();
    }
    int incl = buf[threadIdx.x];
    int cb = carry;
    if (i < n) {
      starts[i] = cb + incl - v;
      if (i == n - 1) starts[n] = cb + incl;
    }
    __syncthreads();
    if (threadIdx.x == 1023) carry = cb + buf[1023];
    __syncthreads();
  }
}

// 5) per-atom sequence position within its shape (1-based), clamped to pos_emb range
__global__ void atom_pos_kernel(const int* __restrict__ astarts, int* __restrict__ apos,
                                int N_, int S_) {
  int i = blockIdx.x * blockDim.x + threadIdx.x;
  if (i >= N_) return;
  int s = find_seg(astarts, S_, i);
  int p = i - astarts[s] + 1;
  if (p < 0) p = 0;
  if (p > NPOS - 1) p = NPOS - 1;
  apos[i] = p;
}

// ---------------------------------------------------------------------------
// 6) WMMA GEMM: Ps[S,240] = shape_emb[S,256] @ W1[224:480,240]  (+ b1)
//    one wave per 16-row M tile; 15 N-tiles x 8 K-steps of v_wmma_f32_16x16x32_f16
// ---------------------------------------------------------------------------
__global__ __launch_bounds__(128) void gemm_ps_kernel(
    const _Float16* __restrict__ A,    // [S][256] f16
    const _Float16* __restrict__ BT,   // [240][256] f16 (transposed W1 slice)
    const float*    __restrict__ b1,   // [240]
    float*          __restrict__ Ps,   // [S][240]
    int S_) {
  int gwave = (int)((blockIdx.x * blockDim.x + threadIdx.x) >> 5);
  int lane  = (int)(threadIdx.x & 31);
  int m0 = gwave * 16;
  if (m0 >= S_) return;                 // wave-uniform: EXEC stays all-ones
  int l    = lane & 15;
  int half = (lane >> 4) & 1;           // 0: lanes 0-15, 1: lanes 16-31
  const _Float16* arow = A + (size_t)(m0 + l) * HDIM;
  const int aoff = half ? 8  : 0;       // A: K 0-7/16-23 vs 8-15/24-31
  const int boff = half ? 8  : 0;       // B: K 0-7/16-23 vs 8-15/24-31

  v8f acc[15] = {};
  for (int k0 = 0; k0 < HDIM; k0 += 32) {
    v16h a;
    ((v4u*)&a)[0] = *(const v4u*)(arow + k0 + aoff);
    ((v4u*)&a)[1] = *(const v4u*)(arow + k0 + aoff + 16);
    #pragma unroll
    for (int nt = 0; nt < 15; ++nt) {
      const _Float16* brow = BT + (size_t)(nt * 16 + l) * HDIM + k0 + boff;
      v16h b;
      ((v4u*)&b)[0] = *(const v4u*)(brow);
      ((v4u*)&b)[1] = *(const v4u*)(brow + 16);
      acc[nt] = __builtin_amdgcn_wmma_f32_16x16x32_f16(
          /*neg_a=*/false, a, /*neg_b=*/false, b,
          /*c_mod=*/(short)0, acc[nt], /*reuse_a=*/false, /*reuse_b=*/false);
    }
  }
  // C/D layout: VGPR r -> M = m0 + r + 8*half ; N = nt*16 + (lane&15)
  #pragma unroll
  for (int nt = 0; nt < 15; ++nt) {
    union { v8f v; float f[8]; } u; u.v = acc[nt];
    int n = nt * 16 + l;
    float bb = b1[n];
    #pragma unroll
    for (int r = 0; r < 8; ++r) {
      int m = m0 + r + (half ? 8 : 0);
      Ps[(size_t)m * HID + n] = u.f[r] + bb;
    }
  }
}

// ---------------------------------------------------------------------------
// 7) edge kernel: h = relu(sum of 7 gathered 240-vectors); logits = h @ W2 + b2
// ---------------------------------------------------------------------------
__global__ __launch_bounds__(256) void edge_kernel(
    const int* __restrict__ row, const int* __restrict__ col,
    const int* __restrict__ ids, const int* __restrict__ chs, const int* __restrict__ apos,
    const int* __restrict__ estarts, const float* __restrict__ Ps,
    const float* __restrict__ Tr, const float* __restrict__ Tc,
    const float* __restrict__ W2, const float* __restrict__ b2,
    float* __restrict__ logits, int E_, int S_) {
  __shared__ float w2s[HID * 4];
  __shared__ float b2s[4];
  for (int i = threadIdx.x; i < HID * 4; i += blockDim.x) w2s[i] = W2[i];
  if (threadIdx.x < 4) b2s[threadIdx.x] = b2[threadIdx.x];
  __syncthreads();

  int e = blockIdx.x * blockDim.x + threadIdx.x;
  if (e >= E_) return;

  int s = find_seg(estarts, S_, e);
  int r = row[e], c = col[e];

  const float4* p0 = (const float4*)(Ps + (size_t)s * HID);                                   // shape proj (+b1)
  const float4* p1 = (const float4*)(Tr + (size_t)ids[r] * HID);
  const float4* p2 = (const float4*)(Tr + (size_t)(NIDS * HID) + (size_t)chs[r] * HID);
  const float4* p3 = (const float4*)(Tr + (size_t)((NIDS + NCHG) * HID) + (size_t)apos[r] * HID);
  const float4* p4 = (const float4*)(Tc + (size_t)ids[c] * HID);
  const float4* p5 = (const float4*)(Tc + (size_t)(NIDS * HID) + (size_t)chs[c] * HID);
  const float4* p6 = (const float4*)(Tc + (size_t)((NIDS + NCHG) * HID) + (size_t)apos[c] * HID);

  float a0 = 0.f, a1 = 0.f, a2 = 0.f, a3 = 0.f;
  #pragma unroll 4
  for (int j4 = 0; j4 < HID / 4; ++j4) {
    float4 x0 = p0[j4], x1 = p1[j4], x2 = p2[j4], x3 = p3[j4],
           x4 = p4[j4], x5 = p5[j4], x6 = p6[j4];
    float h0 = x0.x + x1.x + x2.x + x3.x + x4.x + x5.x + x6.x;
    float h1 = x0.y + x1.y + x2.y + x3.y + x4.y + x5.y + x6.y;
    float h2 = x0.z + x1.z + x2.z + x3.z + x4.z + x5.z + x6.z;
    float h3 = x0.w + x1.w + x2.w + x3.w + x4.w + x5.w + x6.w;
    h0 = h0 > 0.f ? h0 : 0.f;
    h1 = h1 > 0.f ? h1 : 0.f;
    h2 = h2 > 0.f ? h2 : 0.f;
    h3 = h3 > 0.f ? h3 : 0.f;
    int jb = j4 * 16;
    a0 += h0 * w2s[jb + 0] + h1 * w2s[jb + 4] + h2 * w2s[jb + 8]  + h3 * w2s[jb + 12];
    a1 += h0 * w2s[jb + 1] + h1 * w2s[jb + 5] + h2 * w2s[jb + 9]  + h3 * w2s[jb + 13];
    a2 += h0 * w2s[jb + 2] + h1 * w2s[jb + 6] + h2 * w2s[jb + 10] + h3 * w2s[jb + 14];
    a3 += h0 * w2s[jb + 3] + h1 * w2s[jb + 7] + h2 * w2s[jb + 11] + h3 * w2s[jb + 15];
  }
  float4 o;
  o.x = a0 + b2s[0]; o.y = a1 + b2s[1]; o.z = a2 + b2s[2]; o.w = a3 + b2s[3];
  ((float4*)logits)[e] = o;
}

// 8) symmetrize: out[e] = 0.5*(logits[e] + logits[rev[e]])
__global__ void sym_kernel(const float* __restrict__ L, const int* __restrict__ rev,
                           float* __restrict__ out, int E_) {
  int e = blockIdx.x * blockDim.x + threadIdx.x;
  if (e >= E_) return;
  float4 a = ((const float4*)L)[e];
  float4 b = ((const float4*)L)[rev[e]];
  float4 o;
  o.x = 0.5f * (a.x + b.x);
  o.y = 0.5f * (a.y + b.y);
  o.z = 0.5f * (a.z + b.z);
  o.w = 0.5f * (a.w + b.w);
  ((float4*)out)[e] = o;
}

// ---------------------------------------------------------------------------
// host launcher
// ---------------------------------------------------------------------------
extern "C" void kernel_launch(void* const* d_in, const int* in_sizes, int n_in,
                              void* d_out, int out_size, void* d_ws, size_t ws_size,
                              hipStream_t stream) {
  const float* shape_emb = (const float*)d_in[0];
  const float* id_emb    = (const float*)d_in[1];
  const float* ch_emb    = (const float*)d_in[2];
  const float* pos_emb   = (const float*)d_in[3];
  const float* W1        = (const float*)d_in[4];
  const float* b1        = (const float*)d_in[5];
  const float* W2        = (const float*)d_in[6];
  const float* b2        = (const float*)d_in[7];
  const int*   atom_ids  = (const int*)d_in[8];
  const int*   atom_chg  = (const int*)d_in[9];
  const int*   nnodes    = (const int*)d_in[10];
  const int*   eper      = (const int*)d_in[11];
  const int*   row       = (const int*)d_in[12];
  const int*   col       = (const int*)d_in[13];
  const int*   rev       = (const int*)d_in[14];

  const int S_ = in_sizes[10];
  const int N_ = in_sizes[8];
  const int E_ = in_sizes[12];

  // workspace carve-out (256B aligned slices)
  char* wsb = (char*)d_ws;
  size_t off = 0;
  auto carve = [&](size_t bytes) -> char* {
    char* p = wsb + off;
    off = (off + bytes + 255) & ~(size_t)255;
    return p;
  };
  _Float16* Abf     = (_Float16*)carve((size_t)S_ * HDIM * sizeof(_Float16));
  _Float16* BT      = (_Float16*)carve((size_t)HID * HDIM * sizeof(_Float16));
  float*    Tr      = (float*)carve((size_t)NTAB * HID * sizeof(float));
  float*    Tc      = (float*)carve((size_t)NTAB * HID * sizeof(float));
  int*      astarts = (int*)carve((size_t)(S_ + 1) * sizeof(int));
  int*      estarts = (int*)carve((size_t)(S_ + 1) * sizeof(int));
  int*      apos    = (int*)carve((size_t)N_ * sizeof(int));
  float*    Ps      = (float*)carve((size_t)S_ * HID * sizeof(float));
  float*    logits  = (float*)carve((size_t)E_ * 4 * sizeof(float));
  (void)ws_size; (void)n_in; (void)out_size;

  // 1) shape embeddings -> f16
  {
    int n = S_ * HDIM;
    f32_to_f16_kernel<<<(n + 255) / 256, 256, 0, stream>>>(shape_emb, Abf, n);
  }
  // 2) transposed f16 W1 slice
  build_bt_kernel<<<(HID * HDIM + 255) / 256, 256, 0, stream>>>(W1, BT);
  // 3) embedding x W1-slice tables
  {
    int n = 2 * NTAB * HID;
    build_tables_kernel<<<(n + 255) / 256, 256, 0, stream>>>(id_emb, ch_emb, pos_emb, W1, Tr, Tc);
  }
  // 4) scans
  scan_exclusive_kernel<<<1, 1024, 0, stream>>>(nnodes, S_, astarts);
  scan_exclusive_kernel<<<1, 1024, 0, stream>>>(eper, S_, estarts);
  // 5) per-atom positions
  atom_pos_kernel<<<(N_ + 255) / 256, 256, 0, stream>>>(astarts, apos, N_, S_);
  // 6) WMMA GEMM for shape projections (+ b1)
  {
    int mtiles = (S_ + 15) / 16;               // waves
    int blocks = (mtiles + 3) / 4;             // 4 waves / 128-thread block
    gemm_ps_kernel<<<blocks, 128, 0, stream>>>(Abf, BT, b1, Ps, S_);
  }
  // 7) per-edge fused MLP
  edge_kernel<<<(E_ + 255) / 256, 256, 0, stream>>>(
      row, col, atom_ids, atom_chg, apos, estarts, Ps, Tr, Tc, W2, b2, logits, E_, S_);
  // 8) symmetrize into output
  sym_kernel<<<(E_ + 255) / 256, 256, 0, stream>>>(logits, rev, (float*)d_out, E_);
}
